// GRU_Uncertainly_3161095930208
// MI455X (gfx1250) — matrix operand order repaired and checked by reading, not compiled, not run on hardware
//
#include <hip/hip_runtime.h>
#include <hip/hip_bf16.h>

// MI455X / gfx1250 persistent VRNN kernel.
// 64 workgroups x 256 threads (8 wave32). Each WG owns a 16-row batch tile,
// loops over T=200 steps, all GEMMs via v_wmma_f32_16x16x32_bf16.
// Weights are bf16-packed in WMMA B-fragment order by a prologue kernel and
// stay L2-resident; B-fragment base pointers are laundered (opaque asm +
// readfirstlane) so the compiler cannot LICM-hoist weight loads out of the
// T loop, and are cast back to address_space(1) so they lower to
// global_load_b128 (LOADcnt only) instead of flat_load (LOADcnt+DScnt).

#define B_ 1024
#define T_ 200
#define I_ 64
#define H_ 128

typedef __attribute__((ext_vector_type(16))) __bf16 v16bf;
typedef __attribute__((ext_vector_type(8)))  __bf16 v8bf;
typedef __attribute__((ext_vector_type(8)))  float  v8f;

typedef const v8bf __attribute__((address_space(1))) *gv8_p;

// ---- packed-weight workspace offsets (bf16 elements) ----
enum : unsigned {
  OFF_WZ = 0,            // 512x128
  OFF_WR = 65536,        // 512x128
  OFF_WH = 131072,       // 512x128 (zero block for h1 rows)
  OFF_WHH = 196608,      // 128x128
  OFF_PHI_X = 212992,    // 64x128
  OFF_ENC = 221184,      // 256x128
  OFF_ENC_MEAN = 253952, // 128x128
  OFF_ENC_STD = 270336,
  OFF_PRIOR = 286720,
  OFF_PRIOR_MEAN = 303104,
  OFF_PRIOR_STD = 319488,
  OFF_PHI_Z = 335872,
  OFF_DEC = 352256,      // 256x128
  OFF_DEC_MEAN = 385024, // 128x64
  OFF_DEC_STD = 393216,  // 128x64
  OFF_H2X = 401408,      // 128x64
  WS_ELEMS = 409600      // *2 bytes = 819200 B
};

__device__ __forceinline__ __bf16 f2bf(float f) { return (__bf16)f; }  // hw cvt, RNE
__device__ __forceinline__ float sigf(float x) { return 1.0f / (1.0f + __expf(-x)); }
__device__ __forceinline__ float spf(float x)  { return x > 20.f ? x : log1pf(__expf(x)); }

// Make a wave-uniform pointer opaque to the optimizer while keeping it in
// SGPRs: bounce the halves through VGPRs (opaque asm) and readfirstlane back.
__device__ __forceinline__ const __bf16* launder(const __bf16* p) {
  unsigned long long u = (unsigned long long)p;
  unsigned lo = (unsigned)u, hi = (unsigned)(u >> 32);
  asm volatile("" : "+v"(lo), "+v"(hi));
  lo = __builtin_amdgcn_readfirstlane(lo);
  hi = __builtin_amdgcn_readfirstlane(hi);
  return (const __bf16*)(((unsigned long long)hi << 32) | (unsigned long long)lo);
}

// ---------------- prologue: f32 -> bf16 pack into WMMA B-fragment order ----
// B 32x16 bf16 fragment: lane L, element e -> K = (L<16?0:16)+e, N = L&15.
struct PackJob { const float* src; int dout; int ntKtot; int k0; int nk; unsigned dstOff; int zero; };
struct PackArgs { PackJob j[31]; };

__global__ void pack_kernel(PackArgs pa, __bf16* __restrict__ dst, float* __restrict__ kslot) {
  if (blockIdx.x == 0 && threadIdx.x == 0) *kslot = 0.f;
  PackJob job = pa.j[blockIdx.x];
  int ntN = job.dout >> 4;
  int total = job.nk * ntN * 512;
  for (int idx = threadIdx.x; idx < total; idx += blockDim.x) {
    int tile = idx >> 9, e512 = idx & 511;
    int kl = tile % job.nk, n = tile / job.nk;
    int lane = e512 >> 4, e = e512 & 15;
    int krow = kl * 32 + ((lane & 16) ? 16 : 0) + e;
    int col = n * 16 + (lane & 15);
    float v = job.zero ? 0.f : job.src[krow * job.dout + col];
    dst[job.dstOff + ((unsigned)(n * job.ntKtot + job.k0 + kl) << 9) + e512] = f2bf(v);
  }
}

// ---------------- WMMA fragment loads ----------------
union AV { v16bf v; v8bf h[2]; };

// A 16x32 bf16: lane L row = L&15; e0..15 -> K=(L<16?0:16)+e
__device__ __forceinline__ v16bf load_a(const __bf16* a, int stride, int lane, int kt) {
  int row = lane & 15;
  int off = (lane & 16) ? 16 : 0;
  const __bf16* base = a + row * stride + kt * 32;
  AV r;
  r.h[0] = *(const v8bf*)(base + off);
  r.h[1] = *(const v8bf*)(base + off + 8);
  return r.v;
}
__device__ __forceinline__ v16bf load_b(const __bf16* bp, int lane) {
  gv8_p t = (gv8_p)(const v8bf*)(bp + lane * 16);
  AV r;
  r.h[0] = t[0];
  r.h[1] = t[1];
  return r.v;
}

#define WMMA_BF16(A, Bf, C) \
  __builtin_amdgcn_wmma_f32_16x16x32_bf16(false, (A), false, (Bf), (short)0, (C), false, false)

__device__ __forceinline__ v8f gemm(const __bf16* aLds, int stride, const __bf16* wPack,
                                    int ntK, int nt, int lane, v8f acc) {
  const __bf16* bp = launder(wPack + (unsigned)(nt * ntK) * 512u);
  if (ntK >= 8) {
    // two independent accumulation chains (even/odd ktiles) to halve the
    // serial WMMA dependency latency
    v8f accB = {};
#pragma unroll
    for (int k = 0; k < ntK; k += 2) {
      v16bf a0 = load_a(aLds, stride, lane, k);
      v16bf b0 = load_b(bp + (unsigned)k * 512u, lane);
      acc = WMMA_BF16(a0, b0, acc);
      v16bf a1 = load_a(aLds, stride, lane, k + 1);
      v16bf b1 = load_b(bp + (unsigned)(k + 1) * 512u, lane);
      accB = WMMA_BF16(a1, b1, accB);
    }
#pragma unroll
    for (int v = 0; v < 8; v++) acc[v] += accB[v];
  } else {
#pragma unroll
    for (int k = 0; k < ntK; k++) {
      v16bf a = load_a(aLds, stride, lane, k);
      v16bf b = load_b(bp + (unsigned)k * 512u, lane);
      acc = WMMA_BF16(a, b, acc);
    }
  }
  return acc;
}
// shared-A dual GEMM (independent WMMA chains for ILP)
__device__ __forceinline__ void gemm2(const __bf16* aLds, int stride, const __bf16* w1,
                                      const __bf16* w2, int ntK, int nt, int lane,
                                      v8f& acc1, v8f& acc2) {
  const __bf16* b1 = launder(w1 + (unsigned)(nt * ntK) * 512u);
  const __bf16* b2 = launder(w2 + (unsigned)(nt * ntK) * 512u);
  if (ntK >= 8) {
    v8f a1B = {}, a2B = {};
#pragma unroll
    for (int k = 0; k < ntK; k += 2) {
      v16bf a0 = load_a(aLds, stride, lane, k);
      acc1 = WMMA_BF16(a0, load_b(b1 + (unsigned)k * 512u, lane), acc1);
      acc2 = WMMA_BF16(a0, load_b(b2 + (unsigned)k * 512u, lane), acc2);
      v16bf a1v = load_a(aLds, stride, lane, k + 1);
      a1B = WMMA_BF16(a1v, load_b(b1 + (unsigned)(k + 1) * 512u, lane), a1B);
      a2B = WMMA_BF16(a1v, load_b(b2 + (unsigned)(k + 1) * 512u, lane), a2B);
    }
#pragma unroll
    for (int v = 0; v < 8; v++) { acc1[v] += a1B[v]; acc2[v] += a2B[v]; }
  } else {
#pragma unroll
    for (int k = 0; k < ntK; k++) {
      v16bf a = load_a(aLds, stride, lane, k);
      acc1 = WMMA_BF16(a, load_b(b1 + (unsigned)k * 512u, lane), acc1);
      acc2 = WMMA_BF16(a, load_b(b2 + (unsigned)k * 512u, lane), acc2);
    }
  }
}

// ---------------- main persistent kernel ----------------
__global__ __launch_bounds__(256) void vrnn_main(
    const float* __restrict__ inp, const float* __restrict__ Mk,
    const float* __restrict__ Dl, const float* __restrict__ Ze,
    const __bf16* __restrict__ W,
    const float* __restrict__ b_z, const float* __restrict__ b_r, const float* __restrict__ b_h,
    const float* __restrict__ bb_phi_x, const float* __restrict__ bb_phi_z,
    const float* __restrict__ bb_prior, const float* __restrict__ bb_prior_mean,
    const float* __restrict__ bb_prior_std, const float* __restrict__ bb_enc,
    const float* __restrict__ bb_enc_mean, const float* __restrict__ bb_enc_std,
    const float* __restrict__ bb_dec, const float* __restrict__ bb_dec_mean,
    const float* __restrict__ bb_dec_std, const float* __restrict__ bb_h2x,
    const float* __restrict__ out_w, const float* __restrict__ out_b,
    float* __restrict__ o_out, float* __restrict__ o_xp, float* __restrict__ o_xu,
    float* __restrict__ o_kld)
{
  __shared__ __align__(16) float h1f[16][128];         // f32 hidden state
  __shared__ __align__(16) float f_dm[16][64];
  __shared__ __align__(16) float f_ds[16][64];
  __shared__ __align__(16) float red[256];
  __shared__ __align__(16) __bf16 A_h1[16][136];       // h1 bf16
  __shared__ __align__(16) __bf16 A_cat[16][520];      // [x1|h1|delta|m|x_un|phi_z]
  __shared__ __align__(16) __bf16 A_ph[16][264];       // [phi_x or phi_z | h1]
  __shared__ __align__(16) __bf16 A_x[16][72];         // raw x input
  __shared__ __align__(16) __bf16 A_t1[16][136];
  __shared__ __align__(16) __bf16 A_t2[16][136];

  const int tid = threadIdx.x, lane = tid & 31, wave = tid >> 5;
  const int r0 = blockIdx.x * 16;
  const int cw = wave * 16 + (lane & 15);
  const int hi8 = (lane & 16) ? 8 : 0;
  const int c64 = ((wave < 4) ? wave : (wave - 4)) * 16 + (lane & 15);

  // hoist biases
  const float bz = b_z[cw], br = b_r[cw], bh = b_h[cw];
  const float bpx = bb_phi_x[cw], bpz = bb_phi_z[cw], bpr = bb_prior[cw];
  const float bprm = bb_prior_mean[cw], bprs = bb_prior_std[cw];
  const float ben = bb_enc[cw], benm = bb_enc_mean[cw], bens = bb_enc_std[cw];
  const float bde = bb_dec[cw];
  const float bdm = bb_dec_mean[c64], bds = bb_dec_std[c64], bh2 = bb_h2x[c64];

  for (int i = tid; i < 16 * 128; i += 256) {
    int r = i >> 7, c = i & 127;
    h1f[r][c] = 0.f;
    __bf16 zb = f2bf(0.f);
    A_h1[r][c] = zb; A_ph[r][128 + c] = zb; A_cat[r][64 + c] = zb;
  }
  float kacc = 0.f;
  __syncthreads();

#pragma clang loop unroll(disable)
  for (int t = 0; t < T_; ++t) {
    // P0: load step inputs (flat 16x64, 4 elems/thread)
    float x1r[4], mr[4], dr[4];
#pragma unroll
    for (int j = 0; j < 4; j++) {
      int idx = tid * 4 + j, r = idx >> 6, c = idx & 63;
      long g = ((long)(r0 + r) * T_ + t) * I_ + c;
      x1r[j] = inp[g]; mr[j] = Mk[g]; dr[j] = 1.f - sigf(Dl[g]);
      A_x[r][c] = f2bf(x1r[j]);
      A_cat[r][192 + c] = f2bf(dr[j]);
      A_cat[r][256 + c] = f2bf(mr[j]);
    }
    __syncthreads();

    // P1: phi_x = relu(x1_raw @ Wpx)
    {
      v8f a = {}; a = gemm(&A_x[0][0], 72, W + OFF_PHI_X, 2, wave, lane, a);
#pragma unroll
      for (int v = 0; v < 8; v++) { float y = a[v] + bpx; A_ph[v + hi8][cw] = f2bf(y > 0.f ? y : 0.f); }
    }
    __syncthreads();

    // P2: enc = relu([phi_x,h1] @ Wenc)
    {
      v8f a = {}; a = gemm(&A_ph[0][0], 264, W + OFF_ENC, 8, wave, lane, a);
#pragma unroll
      for (int v = 0; v < 8; v++) { float y = a[v] + ben; A_t1[v + hi8][cw] = f2bf(y > 0.f ? y : 0.f); }
    }
    __syncthreads();

    // P3: enc_mean/enc_std (kept in registers) + prior
    v8f em, es;
    {
      v8f a1 = {}, a2 = {};
      gemm2(&A_t1[0][0], 136, W + OFF_ENC_MEAN, W + OFF_ENC_STD, 4, wave, lane, a1, a2);
#pragma unroll
      for (int v = 0; v < 8; v++) { em[v] = a1[v] + benm; es[v] = spf(a2[v] + bens); }
      v8f a3 = {}; a3 = gemm(&A_h1[0][0], 136, W + OFF_PRIOR, 4, wave, lane, a3);
#pragma unroll
      for (int v = 0; v < 8; v++) { float y = a3[v] + bpr; A_t2[v + hi8][cw] = f2bf(y > 0.f ? y : 0.f); }
    }
    __syncthreads();

    // P4: prior_mean/std, KLD accumulation, z_latent = em + eps*es -> A_t1
    {
      v8f a1 = {}, a2 = {};
      gemm2(&A_t2[0][0], 136, W + OFF_PRIOR_MEAN, W + OFF_PRIOR_STD, 4, wave, lane, a1, a2);
#pragma unroll
      for (int v = 0; v < 8; v++) {
        float pm = a1[v] + bprm, ps = spf(a2[v] + bprs);
        float e = es[v], m_ = em[v];
        float d = m_ - pm;
        kacc += 0.5f * (2.f * __logf(ps) - 2.f * __logf(e) + (e * e + d * d) / (ps * ps) - 1.f);
        float eps = Ze[((long)t * B_ + r0 + v + hi8) * H_ + cw];
        A_t1[v + hi8][cw] = f2bf(m_ + eps * e);
      }
    }
    __syncthreads();

    // P5: phi_z = relu(z_latent @ Wpz) -> A_ph[:,0:128] and A_cat[:,384:512]
    {
      v8f a = {}; a = gemm(&A_t1[0][0], 136, W + OFF_PHI_Z, 4, wave, lane, a);
#pragma unroll
      for (int v = 0; v < 8; v++) {
        float y = a[v] + bpz; __bf16 h = f2bf(y > 0.f ? y : 0.f);
        A_ph[v + hi8][cw] = h; A_cat[v + hi8][384 + cw] = h;
      }
    }
    __syncthreads();

    // P6: dec = relu([phi_z,h1] @ Wdec)
    {
      v8f a = {}; a = gemm(&A_ph[0][0], 264, W + OFF_DEC, 8, wave, lane, a);
#pragma unroll
      for (int v = 0; v < 8; v++) { float y = a[v] + bde; A_t1[v + hi8][cw] = f2bf(y > 0.f ? y : 0.f); }
    }
    __syncthreads();

    // P7: dec_mean (waves 0-3) | dec_std (waves 4-7), dout=64
    if (wave < 4) {
      v8f a = {}; a = gemm(&A_t1[0][0], 136, W + OFF_DEC_MEAN, 4, wave, lane, a);
#pragma unroll
      for (int v = 0; v < 8; v++) f_dm[v + hi8][c64] = sigf(a[v] + bdm);
    } else {
      v8f a = {}; a = gemm(&A_t1[0][0], 136, W + OFF_DEC_STD, 4, wave - 4, lane, a);
#pragma unroll
      for (int v = 0; v < 8; v++) f_ds[v + hi8][c64] = spf(a[v] + bds);
    }
    __syncthreads();

    // P8: x_un and x1 imputation (flat)
#pragma unroll
    for (int j = 0; j < 4; j++) {
      int idx = tid * 4 + j, r = idx >> 6, c = idx & 63;
      float xu = sigf((1.f - mr[j]) * f_ds[r][c]) - 0.5f;
      long g = ((long)(r0 + r) * T_ + t) * I_ + c;
      o_xu[g] = xu;
      float x1n = (t != 0) ? (mr[j] * x1r[j] + (1.f - mr[j]) * f_dm[r][c] * dr[j] * xu) : x1r[j];
      A_cat[r][c] = f2bf(x1n);
      A_cat[r][320 + c] = f2bf(xu);
    }
    __syncthreads();

    // P9: z and r gates (K=512), rh = r*h1 -> A_t2
    v8f zreg;
    {
      v8f a1 = {}, a2 = {};
      gemm2(&A_cat[0][0], 520, W + OFF_WZ, W + OFF_WR, 16, wave, lane, a1, a2);
#pragma unroll
      for (int v = 0; v < 8; v++) {
        zreg[v] = sigf(a1[v] + bz);
        float rr = sigf(a2[v] + br);
        A_t2[v + hi8][cw] = f2bf(rr * h1f[v + hi8][cw]);
      }
    }
    __syncthreads();

    // P10: h_tilde = tanh(cat @ Wh + rh @ Whh + b_h); h1new in regs
    float h1n[8];
    {
      v8f a = {}; a = gemm(&A_cat[0][0], 520, W + OFF_WH, 16, wave, lane, a);
      a = gemm(&A_t2[0][0], 136, W + OFF_WHH, 4, wave, lane, a);
#pragma unroll
      for (int v = 0; v < 8; v++) {
        float ht = tanhf(a[v] + bh);
        float h0 = h1f[v + hi8][cw];
        h1n[v] = (1.f - zreg[v]) * h0 + zreg[v] * ht;
      }
    }
    __syncthreads();

    // P11: commit h1 to all consumers
#pragma unroll
    for (int v = 0; v < 8; v++) {
      float y = h1n[v];
      h1f[v + hi8][cw] = y;
      __bf16 h = f2bf(y);
      A_h1[v + hi8][cw] = h; A_ph[v + hi8][128 + cw] = h; A_cat[v + hi8][64 + cw] = h;
    }
    __syncthreads();

    // P12: finalx = h1_new @ Wh2x + b -> xpreds (waves 0-3)
    if (wave < 4) {
      v8f a = {}; a = gemm(&A_h1[0][0], 136, W + OFF_H2X, 4, wave, lane, a);
#pragma unroll
      for (int v = 0; v < 8; v++) {
        long g = ((long)(r0 + v + hi8) * T_ + t) * I_ + c64;
        o_xp[g] = a[v] + bh2;
      }
    }
  }

  __syncthreads();
  // out head: sigmoid(h1 @ out_w + out_b), one row per thread 0..15
  if (tid < 16) {
    float s = 0.f;
#pragma unroll 8
    for (int k = 0; k < 128; k++) s += h1f[tid][k] * out_w[k];
    o_out[r0 + tid] = sigf(s + out_b[0]);
  }
  // KLD block reduce + global atomic
  red[tid] = kacc;
  __syncthreads();
  for (int s = 128; s > 0; s >>= 1) {
    if (tid < s) red[tid] += red[tid + s];
    __syncthreads();
  }
  if (tid == 0) atomicAdd(o_kld, red[0]);
}

// ---------------- host launch ----------------
extern "C" void kernel_launch(void* const* d_in, const int* in_sizes, int n_in,
                              void* d_out, int out_size, void* d_ws, size_t ws_size,
                              hipStream_t stream) {
  (void)in_sizes; (void)out_size;
  if (n_in < 51 || ws_size < (size_t)WS_ELEMS * 2) return;
  const float* P[51];
  for (int i = 0; i < 51; i++) P[i] = (const float*)d_in[i];
  // d_in order (setup_inputs dict flattened):
  // 0 input, 1 M, 2 Delta, 3 z_eps,
  // per gate g in z,r,h: w_xg,w_hg,w_dg,w_mg,w_ung,w_zg,b_g  (4..24)
  // then (w,b) pairs: phi_x,phi_z,prior,prior_mean,prior_std,enc,enc_mean,
  //                   enc_std,dec,dec_mean,dec_std,h2x,out   (25..50)
  __bf16* W = (__bf16*)d_ws;
  float* o_out = (float*)d_out;
  float* o_xp = o_out + 1024;
  float* o_xu = o_xp + (size_t)B_ * T_ * I_;
  float* o_kld = o_xu + (size_t)B_ * T_ * I_;

  PackArgs pa{};
  int i = 0;
  auto J = [&](const float* s, int dout, int ntKtot, int k0, int nk, unsigned off, int z) {
    pa.j[i++] = PackJob{s, dout, ntKtot, k0, nk, off, z};
  };
  // fused Wz: [x1 | h1 | delta | m | x_un | phi_z] rows
  J(P[4], 128, 16, 0, 2, OFF_WZ, 0);  J(P[5], 128, 16, 2, 4, OFF_WZ, 0);
  J(P[6], 128, 16, 6, 2, OFF_WZ, 0);  J(P[7], 128, 16, 8, 2, OFF_WZ, 0);
  J(P[8], 128, 16, 10, 2, OFF_WZ, 0); J(P[9], 128, 16, 12, 4, OFF_WZ, 0);
  // fused Wr
  J(P[11], 128, 16, 0, 2, OFF_WR, 0);  J(P[12], 128, 16, 2, 4, OFF_WR, 0);
  J(P[13], 128, 16, 6, 2, OFF_WR, 0);  J(P[14], 128, 16, 8, 2, OFF_WR, 0);
  J(P[15], 128, 16, 10, 2, OFF_WR, 0); J(P[16], 128, 16, 12, 4, OFF_WR, 0);
  // fused Wh (zero block where h1 sits; w_hh handled separately with r*h1)
  J(P[18], 128, 16, 0, 2, OFF_WH, 0);  J(nullptr, 128, 16, 2, 4, OFF_WH, 1);
  J(P[20], 128, 16, 6, 2, OFF_WH, 0);  J(P[21], 128, 16, 8, 2, OFF_WH, 0);
  J(P[22], 128, 16, 10, 2, OFF_WH, 0); J(P[23], 128, 16, 12, 4, OFF_WH, 0);
  J(P[19], 128, 4, 0, 4, OFF_WHH, 0);
  J(P[25], 128, 2, 0, 2, OFF_PHI_X, 0);
  J(P[35], 128, 8, 0, 8, OFF_ENC, 0);
  J(P[37], 128, 4, 0, 4, OFF_ENC_MEAN, 0);
  J(P[39], 128, 4, 0, 4, OFF_ENC_STD, 0);
  J(P[29], 128, 4, 0, 4, OFF_PRIOR, 0);
  J(P[31], 128, 4, 0, 4, OFF_PRIOR_MEAN, 0);
  J(P[33], 128, 4, 0, 4, OFF_PRIOR_STD, 0);
  J(P[27], 128, 4, 0, 4, OFF_PHI_Z, 0);
  J(P[41], 128, 8, 0, 8, OFF_DEC, 0);
  J(P[43], 64, 4, 0, 4, OFF_DEC_MEAN, 0);
  J(P[45], 64, 4, 0, 4, OFF_DEC_STD, 0);
  J(P[47], 64, 4, 0, 4, OFF_H2X, 0);

  pack_kernel<<<31, 256, 0, stream>>>(pa, W, o_kld);
  vrnn_main<<<B_ / 16, 256, 0, stream>>>(
      P[0], P[1], P[2], P[3], W,
      P[10], P[17], P[24],
      P[26], P[28], P[30], P[32], P[34], P[36], P[38], P[40], P[42], P[44], P[46], P[48],
      P[49], P[50],
      o_out, o_xp, o_xu, o_kld);
}
